// CausalMHSA_21440476741891
// MI455X (gfx1250) — matrix operand order repaired and checked by reading, not compiled, not run on hardware
//
#include <hip/hip_runtime.h>

#define DMODEL 1024
#define NH 16
#define HD 64
#define SEQ 2048
#define BATCH 2

typedef unsigned short u16;
typedef __attribute__((ext_vector_type(16))) __bf16    v16bf;
typedef __attribute__((ext_vector_type(8)))  float     v8f;
typedef __attribute__((ext_vector_type(4)))  unsigned  uivec4;
typedef __attribute__((ext_vector_type(2)))  unsigned  uivec2;
typedef __attribute__((ext_vector_type(4)))  float     fvec4;
typedef __attribute__((ext_vector_type(8)))  int       ivec8;
typedef __attribute__((ext_vector_type(4)))  int       ivec4;

#if defined(__has_builtin)
#if __has_builtin(__builtin_amdgcn_tensor_load_to_lds)
#define HAVE_TDM 1
#endif
#endif
#ifndef HAVE_TDM
#define HAVE_TDM 0
#endif

union Frag16 { v16bf v; uivec4 q[2]; unsigned u[8]; };

__device__ __forceinline__ u16 f32_to_bf16(float f) {
  unsigned u = __float_as_uint(f);
  u += 0x7FFFu + ((u >> 16) & 1u);     // round-to-nearest-even
  return (u16)(u >> 16);
}

// Per-lane async global->LDS copy, 16B per lane. Tracked by ASYNCcnt.
__device__ __forceinline__ void async_load_b128(unsigned lds_off, const void* gptr) {
  unsigned long long ga = (unsigned long long)(uintptr_t)gptr;
  asm volatile("global_load_async_to_lds_b128 %0, %1, off"
               :: "v"(lds_off), "v"(ga) : "memory");
}
__device__ __forceinline__ void wait_asynccnt0() {
  asm volatile("s_wait_asynccnt 0x0" ::: "memory");
}

#if HAVE_TDM
// TDM: DMA a 128-row x 32-col bf16 tile (row stride 1024 elems) from global into
// LDS with HW padding producing a 96-byte LDS row stride (16 DW data + 8 DW pad).
// clang-23 builtin arity: (v4u g0, v8i g1, v4i g2, v4i g3, v8i extra, i32 cpol).
__device__ __forceinline__ void tdm_load_tile_a(unsigned lds_off, const u16* gptr) {
  unsigned long long ga = (unsigned long long)(uintptr_t)gptr;
  uivec4 g0;
  g0.x = 1u;                                             // count=1, user mode
  g0.y = lds_off;                                        // lds_addr (bytes)
  g0.z = (unsigned)(ga & 0xFFFFFFFFu);                   // global_addr[31:0]
  g0.w = (unsigned)((ga >> 32) & 0x01FFFFFFu) | (2u << 30); // addr[56:32] | type=2
  ivec8 g1;
  g1[0] = (int)((1u << 16)       // data_size = 2 bytes
              | (1u << 20)       // pad_enable
              | (3u << 22)       // pad_interval: 16 DWORDs (64B) per row
              | (7u << 25));     // pad_amount: 8 DWORDs (32B) -> 96B stride
  g1[1] = (int)(1024u << 16);    // tensor_dim0 = 1024 (K)
  g1[2] = (int)(4096u << 16);    // tensor_dim1 = 4096 rows
  g1[3] = (int)(32u << 16);      // tile_dim0 = 32
  g1[4] = 128;                   // tile_dim1 = 128, tile_dim2 = 0
  g1[5] = 1024;                  // tensor_dim0_stride = 1024 elems
  g1[6] = 0;
  g1[7] = 0;
  ivec4 gz4 = {0, 0, 0, 0};
  ivec8 gz8 = {0, 0, 0, 0, 0, 0, 0, 0};
  __builtin_amdgcn_tensor_load_to_lds(g0, g1, gz4, gz4, gz8, 0);
}
#endif

// A-operand fragment (16x32 bf16, M x K): lane<16 -> row=lane, K {0-7,16-23};
// lane>=16 -> row=lane-16, K {8-15,24-31}.
__device__ __forceinline__ v16bf load_afrag(const u16* p0, int stride, int lane) {
  const int hf = lane >> 4, lr = lane & 15;
  const u16* p = p0 + (size_t)lr * stride + hf * 8;
  Frag16 f;
  f.q[0] = *(const uivec4*)p;
  f.q[1] = *(const uivec4*)(p + 16);
  return f.v;
}

// B-operand fragment (32x16 bf16, K x N): lane<16 -> col=lane, K {0-7,16-23};
// lane>=16 -> col=lane-16, K {8-15,24-31}. Source K-contiguous per column.
__device__ __forceinline__ v16bf load_bfrag(const u16* p0, int stride, int lane) {
  const int hf = lane >> 4, lr = lane & 15;
  const u16* p = p0 + (size_t)lr * stride + hf * 8;
  Frag16 f;
  f.q[0] = *(const uivec4*)p;
  f.q[1] = *(const uivec4*)(p + 16);
  return f.v;
}

// ---------------- f32 -> bf16 convert (vectorized) ----------------
__global__ __launch_bounds__(256) void cvt_f32_bf16(const float* __restrict__ x,
                                                    u16* __restrict__ y, int n4) {
  for (int i = blockIdx.x * blockDim.x + threadIdx.x; i < n4;
       i += gridDim.x * blockDim.x) {
    fvec4 f = ((const fvec4*)x)[i];
    uivec2 o;
    o.x = ((unsigned)f32_to_bf16(f.y) << 16) | f32_to_bf16(f.x);
    o.y = ((unsigned)f32_to_bf16(f.w) << 16) | f32_to_bf16(f.z);
    ((uivec2*)y)[i] = o;
  }
}

// ---------------- tiled WMMA GEMM: C[m,n] = sum_k A[m,k]*B[n,k] ----------------
// A tile staged by TDM (one D# per iteration, wave 0), B tile by per-lane async
// loads; double-buffered LDS so DMA of tile i+1 overlaps WMMAs of tile i.
// MODE 0: bf16 store to Cbf [M,NTOT]; V third (n>=2048) also transposed into Vt.
// MODE 1: f32 store to Cf.
template <int NTOT, int MODE>
__global__ __launch_bounds__(256) void gemm_bf16_wmma(
    const u16* __restrict__ A, const u16* __restrict__ Bm,
    u16* __restrict__ Cbf, u16* __restrict__ Vt, float* __restrict__ Cf) {
  __shared__ __attribute__((aligned(16))) u16 sA[2][128 * 48];
  __shared__ __attribute__((aligned(16))) u16 sB[2][128 * 48];

  const int tid = threadIdx.x;
  const int lane = tid & 31;
  const int wv = tid >> 5;
  const int bm = blockIdx.y * 128;
  const int bn = blockIdx.x * 128;
  const int wm = (wv & 1) * 64;
  const int wn = (wv >> 1) * 32;

  auto stage = [&](int ib, int kb) {
#pragma unroll
    for (int i = 0; i < 2; i++) {           // B: 512 16B chunks over 256 threads
      int c = tid + i * 256;
      int row = c >> 2;
      int col = (c & 3) * 8;
      async_load_b128((unsigned)(uintptr_t)&sB[ib][row * 48 + col],
                      &Bm[(size_t)(bn + row) * DMODEL + kb + col]);
    }
#if HAVE_TDM
    if (wv == 0)
      tdm_load_tile_a((unsigned)(uintptr_t)&sA[ib][0],
                      &A[(size_t)bm * DMODEL + kb]);
#else
#pragma unroll
    for (int i = 0; i < 2; i++) {
      int c = tid + i * 256;
      int row = c >> 2;
      int col = (c & 3) * 8;
      async_load_b128((unsigned)(uintptr_t)&sA[ib][row * 48 + col],
                      &A[(size_t)(bm + row) * DMODEL + kb + col]);
    }
#endif
  };

  const v8f vz = {0.f, 0.f, 0.f, 0.f, 0.f, 0.f, 0.f, 0.f};
  v8f acc[4][2];
#pragma unroll
  for (int i = 0; i < 4; i++)
#pragma unroll
    for (int j = 0; j < 2; j++) acc[i][j] = vz;

  stage(0, 0);

  int ib = 0;
  for (int kb = 0; kb < DMODEL; kb += 32, ib ^= 1) {
    wait_asynccnt0();
#if HAVE_TDM
    if (wv == 0) __builtin_amdgcn_s_wait_tensorcnt(0);
#endif
    __syncthreads();                         // tile ib ready for all waves
    if (kb + 32 < DMODEL) stage(ib ^ 1, kb + 32);  // overlap next DMA

    v16bf af[4], bfr[2];
#pragma unroll
    for (int i = 0; i < 4; i++)
      af[i] = load_afrag(&sA[ib][(wm + i * 16) * 48], 48, lane);
#pragma unroll
    for (int j = 0; j < 2; j++)
      bfr[j] = load_bfrag(&sB[ib][(wn + j * 16) * 48], 48, lane);
#pragma unroll
    for (int i = 0; i < 4; i++)
#pragma unroll
      for (int j = 0; j < 2; j++)
        acc[i][j] = __builtin_amdgcn_wmma_f32_16x16x32_bf16(
            false, af[i], false, bfr[j], (short)0, acc[i][j], false, false);
  }

  const int hf = lane >> 4, lr = lane & 15;
#pragma unroll
  for (int i = 0; i < 4; i++)
#pragma unroll
    for (int j = 0; j < 2; j++)
#pragma unroll
      for (int r = 0; r < 8; r++) {
        int n = bn + wn + j * 16 + lr;
        int m = bm + wm + i * 16 + r + hf * 8;
        float val = acc[i][j][r];
        if (MODE == 0) {
          u16 hv = f32_to_bf16(val);
          Cbf[(size_t)m * NTOT + n] = hv;
          if (n >= 2 * DMODEL) {             // V columns -> transposed copy
            int e = n - 2 * DMODEL;
            int hh = e >> 6, d = e & 63;
            int b_ = m >> 11, s_ = m & (SEQ - 1);
            Vt[(((size_t)b_ * NH + hh) * HD + d) * SEQ + s_] = hv;
          }
        } else {
          Cf[(size_t)m * NTOT + n] = val;
        }
      }
}

// ---------------- flash attention, one wave per 16 query rows ----------------
// S^T = K*Q^T so D-layout lane == query; softmax row reduce = per-lane VGPR
// reduce + one shfl_xor(16); packed S^T tiles land in A-operand layout for P*V.
// Pipelined: K frags double-buffered (prefetch kb+32 during softmax), V frags
// issued at the top of the iteration.
__global__ __launch_bounds__(256) void attn_wmma(const u16* __restrict__ QKV,
                                                 const u16* __restrict__ Vt,
                                                 u16* __restrict__ CTX) {
  const int tid = threadIdx.x;
  const int lane = tid & 31;
  const int wv = tid >> 5;
  const int hf = lane >> 4, lr = lane & 15;
  const int b = blockIdx.z, h = blockIdx.y;
  const int qb = blockIdx.x * 128 + wv * 16;

  const u16* Qbase = QKV + (size_t)(b * SEQ + qb) * (3 * DMODEL) + h * HD;
  const u16* Kcol = QKV + (size_t)b * SEQ * (3 * DMODEL) + DMODEL + h * HD;
  const u16* Vbase = Vt + ((size_t)(b * NH + h) * HD) * SEQ;

  v16bf qf0 = load_bfrag(Qbase + 0, 3 * DMODEL, lane);   // B-op: [d, q] d 0..31
  v16bf qf1 = load_bfrag(Qbase + 32, 3 * DMODEL, lane);  // d 32..63

  const v8f vz = {0.f, 0.f, 0.f, 0.f, 0.f, 0.f, 0.f, 0.f};
  v8f O[4] = {vz, vz, vz, vz};
  float m_run = -1e30f, l_run = 0.f;
  const float scale = 0.125f;  // 1/sqrt(64)
  const int qg = qb + lr;
  const int kend = qb + 16;

  // preload K fragments for kb = 0
  v16bf c0a = load_afrag(Kcol, 3 * DMODEL, lane);
  v16bf c0b = load_afrag(Kcol + 32, 3 * DMODEL, lane);
  v16bf c1a = load_afrag(Kcol + (size_t)16 * (3 * DMODEL), 3 * DMODEL, lane);
  v16bf c1b = load_afrag(Kcol + (size_t)16 * (3 * DMODEL) + 32, 3 * DMODEL, lane);

  for (int kb = 0; kb < kend; kb += 32) {
    // V fragments for this chunk, issued early (consumed after softmax)
    v16bf vf[4];
#pragma unroll
    for (int nt = 0; nt < 4; nt++)
      vf[nt] = load_bfrag(Vbase + (size_t)(nt * 16) * SEQ + kb, SEQ, lane);

    v8f t0 = vz, t1 = vz;  // S^T tiles: M=key, N=query
    t0 = __builtin_amdgcn_wmma_f32_16x16x32_bf16(false, c0a, false, qf0, (short)0, t0, false, false);
    t0 = __builtin_amdgcn_wmma_f32_16x16x32_bf16(false, c0b, false, qf1, (short)0, t0, false, false);
    t1 = __builtin_amdgcn_wmma_f32_16x16x32_bf16(false, c1a, false, qf0, (short)0, t1, false, false);
    t1 = __builtin_amdgcn_wmma_f32_16x16x32_bf16(false, c1b, false, qf1, (short)0, t1, false, false);

    // prefetch next chunk's K fragments (latency hidden by softmax below)
    const int kn = kb + 32;
    if (kn < kend) {
      const u16* kp = Kcol + (size_t)kn * (3 * DMODEL);
      c0a = load_afrag(kp, 3 * DMODEL, lane);
      c0b = load_afrag(kp + 32, 3 * DMODEL, lane);
      c1a = load_afrag(kp + (size_t)16 * (3 * DMODEL), 3 * DMODEL, lane);
      c1b = load_afrag(kp + (size_t)16 * (3 * DMODEL) + 32, 3 * DMODEL, lane);
    }

    float cm = -1e30f;
#pragma unroll
    for (int r = 0; r < 8; r++) {
      int key0 = kb + r + hf * 8;
      int key1 = key0 + 16;
      float s0 = t0[r] * scale; if (key0 > qg) s0 = -1e30f;
      float s1 = t1[r] * scale; if (key1 > qg) s1 = -1e30f;
      t0[r] = s0; t1[r] = s1;
      cm = fmaxf(cm, fmaxf(s0, s1));
    }
    cm = fmaxf(cm, __shfl_xor(cm, 16, 32));
    float mnew = fmaxf(m_run, cm);
    float alpha = __expf(m_run - mnew);
    l_run *= alpha;
#pragma unroll
    for (int nt = 0; nt < 4; nt++)
#pragma unroll
      for (int r = 0; r < 8; r++) O[nt][r] = O[nt][r] * alpha;

    float rs = 0.f;
#pragma unroll
    for (int r = 0; r < 8; r++) { float p = __expf(t0[r] - mnew); rs += p; t0[r] = p; }
#pragma unroll
    for (int r = 0; r < 8; r++) { float p = __expf(t1[r] - mnew); rs += p; t1[r] = p; }
    rs += __shfl_xor(rs, 16, 32);
    l_run += rs;
    m_run = mnew;

    // Pack P (S^T layout) into A-operand bf16 fragment for P*V (K = 32 keys)
    Frag16 pf;
#pragma unroll
    for (int j = 0; j < 4; j++) {
      pf.u[j]     = ((unsigned)f32_to_bf16(t0[2 * j + 1]) << 16) | f32_to_bf16(t0[2 * j]);
      pf.u[4 + j] = ((unsigned)f32_to_bf16(t1[2 * j + 1]) << 16) | f32_to_bf16(t1[2 * j]);
    }

#pragma unroll
    for (int nt = 0; nt < 4; nt++)
      O[nt] = __builtin_amdgcn_wmma_f32_16x16x32_bf16(
          false, pf.v, false, vf[nt], (short)0, O[nt], false, false);
  }

#pragma unroll
  for (int nt = 0; nt < 4; nt++)
#pragma unroll
    for (int r = 0; r < 8; r++) {
      int qoff = r + hf * 8;
      float lv = __shfl(l_run, qoff, 32);
      float val = O[nt][r] / lv;
      int q = qb + qoff;
      int d = h * HD + nt * 16 + lr;
      CTX[(size_t)(b * SEQ + q) * DMODEL + d] = f32_to_bf16(val);
    }
}

// ---------------- launch ----------------
extern "C" void kernel_launch(void* const* d_in, const int* in_sizes, int n_in,
                              void* d_out, int out_size, void* d_ws, size_t ws_size,
                              hipStream_t stream) {
  (void)in_sizes; (void)n_in; (void)out_size; (void)ws_size;
  const float* x = (const float*)d_in[0];     // [2,2048,1024]
  const float* wqkv = (const float*)d_in[1];  // [3072,1024]
  const float* wo = (const float*)d_in[2];    // [1024,1024]
  float* out = (float*)d_out;                 // [2,2048,1024]

  char* ws = (char*)d_ws;
  size_t off = 0;
  auto carve = [&](size_t bytes) {
    size_t p = off;
    off += (bytes + 255) & ~(size_t)255;
    return p;
  };
  u16* Xb  = (u16*)(ws + carve((size_t)BATCH * SEQ * DMODEL * 2));
  u16* Wqb = (u16*)(ws + carve((size_t)3 * DMODEL * DMODEL * 2));
  u16* Wob = (u16*)(ws + carve((size_t)DMODEL * DMODEL * 2));
  u16* QKV = (u16*)(ws + carve((size_t)BATCH * SEQ * 3 * DMODEL * 2));
  u16* Vt  = (u16*)(ws + carve((size_t)BATCH * SEQ * DMODEL * 2));
  u16* CTX = (u16*)(ws + carve((size_t)BATCH * SEQ * DMODEL * 2));

  cvt_f32_bf16<<<256, 256, 0, stream>>>(x, Xb, BATCH * SEQ * DMODEL / 4);
  cvt_f32_bf16<<<256, 256, 0, stream>>>(wqkv, Wqb, 3 * DMODEL * DMODEL / 4);
  cvt_f32_bf16<<<64, 256, 0, stream>>>(wo, Wob, DMODEL * DMODEL / 4);

  gemm_bf16_wmma<3 * DMODEL, 0><<<dim3(24, 32), 256, 0, stream>>>(
      Xb, Wqb, QKV, Vt, nullptr);

  attn_wmma<<<dim3(SEQ / 128, NH, BATCH), 256, 0, stream>>>(QKV, Vt, CTX);

  gemm_bf16_wmma<DMODEL, 1><<<dim3(8, 32), 256, 0, stream>>>(
      CTX, Wob, nullptr, nullptr, out);
}
